// LanguagePlanningHeadLayer_79585743995164
// MI455X (gfx1250) — matrix operand order repaired and checked by reading, not compiled, not run on hardware
//
#include <hip/hip_runtime.h>
#include <hip/hip_bf16.h>

typedef __attribute__((ext_vector_type(16))) __bf16 v16bf;
typedef __attribute__((ext_vector_type(8)))  __bf16 v8bf;
typedef __attribute__((ext_vector_type(8)))  float  v8f;

#define NB       4
#define HEADS    8
#define DHEAD    64
#define DINNER   512
#define LLM_EMB  4096
#define NPLAN    1024
#define SVLM     512
#define ROWS     (NB * NPLAN)     // 4096
#define KLEN     (SVLM + NPLAN)   // 1536

__device__ __forceinline__ __bf16 f2bf(float f) {
    unsigned u = __builtin_bit_cast(unsigned, f);
    unsigned r = u + 0x7FFFu + ((u >> 16) & 1u);
    unsigned short h = (unsigned short)(r >> 16);
    return __builtin_bit_cast(__bf16, h);
}

// ---------------- elementwise / prep kernels ----------------

// Wt[n*K + k] = bf16(W[k*N + n])   (W is K x N row-major)
__global__ void k_transpose_bf16(const float* __restrict__ W, __bf16* __restrict__ Wt,
                                 int K, int N) {
    int idx = blockIdx.x * 256 + threadIdx.x;
    if (idx >= K * N) return;
    int k = idx / N, n = idx % N;
    Wt[(size_t)n * K + k] = f2bf(W[idx]);
}

// X = concat(vlm_keys, vlm_values, axis=1) -> bf16, (ROWS x LLM_EMB)
__global__ void k_build_xskv(const float* __restrict__ keys, const float* __restrict__ vals,
                             __bf16* __restrict__ out) {
    size_t idx = (size_t)blockIdx.x * 256 + threadIdx.x;   // ROWS*LLM_EMB = 16.7M
    int c = (int)(idx & (LLM_EMB - 1));
    int r = (int)(idx >> 12);
    int bb = r >> 10, t = r & 1023;
    float v = (t < SVLM) ? keys[((size_t)(bb * SVLM + t)) * LLM_EMB + c]
                         : vals[((size_t)(bb * SVLM + (t - SVLM))) * LLM_EMB + c];
    out[idx] = f2bf(v);
}

// per-row RMSNorm over 512, out bf16. block=256, each thread 2 elems.
__global__ void k_rmsnorm_bf16(const float* __restrict__ x, const float* __restrict__ g,
                               __bf16* __restrict__ out) {
    int row = blockIdx.x, t = threadIdx.x;
    const float* xr = x + (size_t)row * DINNER;
    float x0 = xr[t], x1 = xr[t + 256];
    __shared__ float red[256];
    red[t] = x0 * x0 + x1 * x1;
    __syncthreads();
    for (int s = 128; s > 0; s >>= 1) {
        if (t < s) red[t] += red[t + s];
        __syncthreads();
    }
    float rs = rsqrtf(red[0] * (1.0f / DINNER) + 1e-6f);
    out[(size_t)row * DINNER + t]       = f2bf(x0 * rs * (1.0f + g[t]));
    out[(size_t)row * DINNER + t + 256] = f2bf(x1 * rs * (1.0f + g[t + 256]));
}

// gelu(exact) fp32 -> bf16
__global__ void k_gelu_bf16(const float* __restrict__ in, __bf16* __restrict__ out) {
    size_t idx = (size_t)blockIdx.x * 256 + threadIdx.x;
    float x = in[idx];
    out[idx] = f2bf(0.5f * x * (1.0f + erff(x * 0.70710678118654752f)));
}

// RoPE on q and self-k; q -> (b,h,n,64) bf16 ; self k -> Kbf rows [512..1536)
__global__ void k_rope_qk(const float* __restrict__ qkv, __bf16* __restrict__ qbf,
                          __bf16* __restrict__ Kbf) {
    int idx = blockIdx.x * 256 + threadIdx.x;   // NB*HEADS*NPLAN*32
    int d  = idx & 31;
    int i  = (idx >> 5) & (NPLAN - 1);
    int h  = (idx >> 15) & (HEADS - 1);
    int bb = idx >> 18;
    size_t row = (size_t)(bb * NPLAN + i) * (3 * DINNER);
    // inv_freq = 10000^{-d/32} = exp2(-d * log2(10000)/32)
    float f = (float)i * exp2f(-(float)d * (13.287712379549449f / 32.0f));
    float c = cosf(f), s = sinf(f);
    int cq = h * DHEAD + d;
    float q1 = qkv[row + cq], q2 = qkv[row + cq + 32];
    size_t qo = (size_t)((bb * HEADS + h) * NPLAN + i) * DHEAD + d;
    qbf[qo]      = f2bf(q1 * c - q2 * s);
    qbf[qo + 32] = f2bf(q2 * c + q1 * s);
    int ck = DINNER + h * DHEAD + d;
    float k1 = qkv[row + ck], k2 = qkv[row + ck + 32];
    size_t ko = (size_t)((bb * HEADS + h) * KLEN + SVLM + i) * DHEAD + d;
    Kbf[ko]      = f2bf(k1 * c - k2 * s);
    Kbf[ko + 32] = f2bf(k2 * c + k1 * s);
}

// state keys (no rope) -> Kbf rows [0..512)
__global__ void k_fill_state_keys(const float* __restrict__ statekv, __bf16* __restrict__ Kbf) {
    int idx = blockIdx.x * 256 + threadIdx.x;   // NB*HEADS*SVLM*64
    int d  = idx & 63;
    int j  = (idx >> 6) & (SVLM - 1);
    int h  = (idx >> 15) & (HEADS - 1);
    int bb = idx >> 18;
    float v = statekv[(size_t)(bb * NPLAN + j) * DINNER + h * DHEAD + d];
    Kbf[(size_t)((bb * HEADS + h) * KLEN + j) * DHEAD + d] = f2bf(v);
}

// values (state + self), stored d-major: Vt[(bh*64+d)*1536 + j]
__global__ void k_fill_vt(const float* __restrict__ statekv, const float* __restrict__ qkv,
                          __bf16* __restrict__ Vt) {
    int idx = blockIdx.x * 256 + threadIdx.x;   // NB*HEADS*KLEN*64
    int d    = idx & 63;
    int rest = idx >> 6;
    int j    = rest % KLEN;
    int bh   = rest / KLEN;
    int h = bh & (HEADS - 1), bb = bh >> 3;
    float v;
    if (j < SVLM)
        v = statekv[(size_t)(bb * NPLAN + SVLM + j) * DINNER + h * DHEAD + d];
    else
        v = qkv[(size_t)(bb * NPLAN + (j - SVLM)) * (3 * DINNER) + 2 * DINNER + h * DHEAD + d];
    Vt[(size_t)(bh * DHEAD + d) * KLEN + j] = f2bf(v);
}

// ---------------- register-blocked bf16 WMMA GEMM ----------------
// C[M x N] (fp32) = A[M x K](bf16,row-major) * Bt[N x K](bf16) + bias[N] (+ resid[M x N])
// Each wave computes a 64x32 output block: 4x2 grid of 16x16 WMMA tiles.
// 8 C-frags (64 VGPRs) + 4 A-frags (32) + 2 B-frags (16); 8 WMMA per 12 b128 loads.
// waves_per_eu(1) lets the allocator use the full VGPR file (no spills).
// block = 128 threads (4 waves in 2x2) => 128x64 block tile; grid = (N/64, M/128)
__global__ __attribute__((amdgpu_waves_per_eu(1)))
void k_wmma_gemm(const __bf16* __restrict__ A, const __bf16* __restrict__ Bt,
                 const float* __restrict__ bias, const float* __restrict__ resid,
                 float* __restrict__ C, int M, int N, int K) {
    int lane = threadIdx.x & 31;
    int w    = threadIdx.x >> 5;
    int mc   = blockIdx.y * 128 + (w & 1) * 64;
    int nc   = blockIdx.x * 64 + (w >> 1) * 32;
    int half = lane >> 4;
    int nl   = lane & 15;

    const __bf16* arow = A + (size_t)(mc + nl) * K + half * 16;
    const __bf16* brow = Bt + (size_t)(nc + nl) * K + half * 16;

    v8f c[4][2];
#pragma unroll
    for (int rt = 0; rt < 4; ++rt)
#pragma unroll
        for (int ct = 0; ct < 2; ++ct) c[rt][ct] = (v8f){};

    for (int kc = 0; kc < K; kc += 32) {
        v16bf a[4], b[2];
#pragma unroll
        for (int rt = 0; rt < 4; ++rt) {
            v8bf lo = *(const v8bf*)(arow + rt * 16 * K + kc);
            v8bf hi = *(const v8bf*)(arow + rt * 16 * K + kc + 8);
#pragma unroll
            for (int e = 0; e < 8; ++e) { a[rt][e] = lo[e]; a[rt][e + 8] = hi[e]; }
        }
#pragma unroll
        for (int ct = 0; ct < 2; ++ct)
            b[ct] = *(const v16bf*)(brow + ct * 16 * K + kc);
#pragma unroll
        for (int rt = 0; rt < 4; ++rt)
#pragma unroll
            for (int ct = 0; ct < 2; ++ct)
                c[rt][ct] = __builtin_amdgcn_wmma_f32_16x16x32_bf16(
                    false, a[rt], false, b[ct], (short)0, c[rt][ct], false, false);
    }

#pragma unroll
    for (int ct = 0; ct < 2; ++ct) {
        int col = nc + ct * 16 + nl;
        float bv = bias ? bias[col] : 0.0f;
#pragma unroll
        for (int rt = 0; rt < 4; ++rt) {
#pragma unroll
            for (int r = 0; r < 8; ++r) {
                int m = mc + rt * 16 + half * 8 + r;
                float v = c[rt][ct][r] + bv;
                if (resid) v += resid[(size_t)m * N + col];
                C[(size_t)m * N + col] = v;
            }
        }
    }
}

// ---------------- flash attention ----------------
// one wave (32 threads) per (b,h,16-query tile). grid.x = NB*HEADS*(NPLAN/16) = 2048
__global__ void k_attention(const __bf16* __restrict__ qbf, const __bf16* __restrict__ Kbf,
                            const __bf16* __restrict__ Vt, __bf16* __restrict__ attnbf) {
    int wid   = blockIdx.x;
    int qtile = wid & 63;
    int bh    = wid >> 6;
    int qbase = qtile * 16;
    int lane  = threadIdx.x;
    int half  = lane >> 4;
    int nl    = lane & 15;

    // q A-fragments for the two 32-deep d-chunks
    const __bf16* qrow = qbf + (size_t)(bh * NPLAN + qbase + nl) * DHEAD;
    v16bf aq[2];
#pragma unroll
    for (int dc = 0; dc < 2; ++dc) {
        v8bf lo = *(const v8bf*)(qrow + dc * 32 + half * 16);
        v8bf hi = *(const v8bf*)(qrow + dc * 32 + 8 + half * 16);
#pragma unroll
        for (int e = 0; e < 8; ++e) { aq[dc][e] = lo[e]; aq[dc][e + 8] = hi[e]; }
    }

    v8f o[4];
#pragma unroll
    for (int t = 0; t < 4; ++t) o[t] = (v8f){};
    float mrun[8], lrun[8];
#pragma unroll
    for (int r = 0; r < 8; ++r) { mrun[r] = -3.0e38f; lrun[r] = 0.0f; }

    __shared__ __bf16 pb[16 * 32];

    const __bf16* kbase = Kbf + (size_t)bh * KLEN * DHEAD;
    const __bf16* vbase = Vt + (size_t)bh * DHEAD * KLEN;
    int jend = qbase + 16 + SVLM;   // last allowed key is qbase+15+512

    for (int jc = 0; jc < jend; jc += 32) {
        v8f s0 = {}, s1 = {};
#pragma unroll
        for (int dc = 0; dc < 2; ++dc) {
            v16bf bk0 = *(const v16bf*)(kbase + (size_t)(jc + nl) * DHEAD + dc * 32 + half * 16);
            v16bf bk1 = *(const v16bf*)(kbase + (size_t)(jc + 16 + nl) * DHEAD + dc * 32 + half * 16);
            s0 = __builtin_amdgcn_wmma_f32_16x16x32_bf16(false, aq[dc], false, bk0, (short)0, s0,
                                                         false, false);
            s1 = __builtin_amdgcn_wmma_f32_16x16x32_bf16(false, aq[dc], false, bk1, (short)0, s1,
                                                         false, false);
        }
        // scale + causal mask (key j allowed iff j <= i + 512)
        float p0[8], p1[8], alpha[8];
#pragma unroll
        for (int r = 0; r < 8; ++r) {
            int i = qbase + half * 8 + r;
            float a0 = (jc + nl      <= i + SVLM) ? s0[r] * 0.125f : -3.0e38f;
            float a1 = (jc + 16 + nl <= i + SVLM) ? s1[r] * 0.125f : -3.0e38f;
            float mx = fmaxf(a0, a1);
#pragma unroll
            for (int msk = 1; msk < 16; msk <<= 1) mx = fmaxf(mx, __shfl_xor(mx, msk, 32));
            float mnew = fmaxf(mrun[r], mx);
            alpha[r] = expf(mrun[r] - mnew);
            p0[r] = expf(a0 - mnew);
            p1[r] = expf(a1 - mnew);
            float rs = p0[r] + p1[r];
#pragma unroll
            for (int msk = 1; msk < 16; msk <<= 1) rs += __shfl_xor(rs, msk, 32);
            lrun[r] = alpha[r] * lrun[r] + rs;
            mrun[r] = mnew;
        }
        // rescale O
#pragma unroll
        for (int t = 0; t < 4; ++t)
#pragma unroll
            for (int r = 0; r < 8; ++r) o[t][r] *= alpha[r];
        // P (C-layout) -> LDS -> A-fragment layout
#pragma unroll
        for (int r = 0; r < 8; ++r) {
            pb[(half * 8 + r) * 32 + nl]      = f2bf(p0[r]);
            pb[(half * 8 + r) * 32 + 16 + nl] = f2bf(p1[r]);
        }
        __syncthreads();
        v16bf pa;
#pragma unroll
        for (int e = 0; e < 8; ++e) {
            pa[e]     = pb[nl * 32 + half * 16 + e];
            pa[e + 8] = pb[nl * 32 + 8 + half * 16 + e];
        }
        __syncthreads();
        // O += P @ V  (4 d-tiles of 16)
#pragma unroll
        for (int t = 0; t < 4; ++t) {
            v16bf bv = *(const v16bf*)(vbase + (size_t)(t * 16 + nl) * KLEN + jc + half * 16);
            o[t] = __builtin_amdgcn_wmma_f32_16x16x32_bf16(false, pa, false, bv, (short)0, o[t],
                                                           false, false);
        }
    }

    int h = bh & (HEADS - 1), bb = bh >> 3;
#pragma unroll
    for (int r = 0; r < 8; ++r) {
        float inv = 1.0f / lrun[r];
        size_t row = (size_t)(bb * NPLAN + qbase + half * 8 + r) * DINNER;
#pragma unroll
        for (int t = 0; t < 4; ++t)
            attnbf[row + h * DHEAD + t * 16 + nl] = f2bf(o[t][r] * inv);
    }
}

// ---------------- host ----------------

extern "C" void kernel_launch(void* const* d_in, const int* in_sizes, int n_in,
                              void* d_out, int out_size, void* d_ws, size_t ws_size,
                              hipStream_t stream) {
    (void)in_sizes; (void)n_in; (void)out_size; (void)ws_size;
    const float* planning = (const float*)d_in[0];
    const float* vlm_keys = (const float*)d_in[1];
    const float* vlm_vals = (const float*)d_in[2];
    const float* W_skv    = (const float*)d_in[3];
    const float* b_skv    = (const float*)d_in[4];
    const float* W_qkv    = (const float*)d_in[5];
    const float* b_qkv    = (const float*)d_in[6];
    const float* W_out    = (const float*)d_in[7];
    const float* b_out    = (const float*)d_in[8];
    const float* W_ff1    = (const float*)d_in[9];
    const float* b_ff1    = (const float*)d_in[10];
    const float* W_ff2    = (const float*)d_in[11];
    const float* b_ff2    = (const float*)d_in[12];
    const float* g_norm1  = (const float*)d_in[13];
    const float* g_norm2  = (const float*)d_in[14];
    float* out = (float*)d_out;

    char* ws = (char*)d_ws;
    size_t o = 0;
    auto take = [&](size_t bytes) { char* p = ws + o; o += (bytes + 255) & ~(size_t)255; return p; };
    __bf16* Xskv    = (__bf16*)take((size_t)ROWS * LLM_EMB * 2);
    __bf16* Wskv_t  = (__bf16*)take((size_t)DINNER * LLM_EMB * 2);
    __bf16* Wqkv_t  = (__bf16*)take((size_t)3 * DINNER * DINNER * 2);
    __bf16* Wout_t  = (__bf16*)take((size_t)DINNER * DINNER * 2);
    __bf16* Wff1_t  = (__bf16*)take((size_t)DINNER * DINNER * 2);
    __bf16* Wff2_t  = (__bf16*)take((size_t)DINNER * DINNER * 2);
    __bf16* normed1 = (__bf16*)take((size_t)ROWS * DINNER * 2);
    float*  statekv = (float*) take((size_t)ROWS * DINNER * 4);
    float*  qkvf    = (float*) take((size_t)ROWS * 3 * DINNER * 4);
    __bf16* qbf     = (__bf16*)take((size_t)NB * HEADS * NPLAN * DHEAD * 2);
    __bf16* Kbf     = (__bf16*)take((size_t)NB * HEADS * KLEN * DHEAD * 2);
    __bf16* Vtbf    = (__bf16*)take((size_t)NB * HEADS * KLEN * DHEAD * 2);
    __bf16* attnbf  = (__bf16*)take((size_t)ROWS * DINNER * 2);
    float*  allout  = (float*) take((size_t)ROWS * DINNER * 4);
    __bf16* normed2 = (__bf16*)take((size_t)ROWS * DINNER * 2);
    float*  ffn1f   = (float*) take((size_t)ROWS * DINNER * 4);
    __bf16* ffn1bf  = (__bf16*)take((size_t)ROWS * DINNER * 2);

    // weight transposes -> bf16 N-major
    k_transpose_bf16<<<(LLM_EMB * DINNER + 255) / 256, 256, 0, stream>>>(W_skv, Wskv_t, LLM_EMB, DINNER);
    k_transpose_bf16<<<(DINNER * 3 * DINNER + 255) / 256, 256, 0, stream>>>(W_qkv, Wqkv_t, DINNER, 3 * DINNER);
    k_transpose_bf16<<<(DINNER * DINNER + 255) / 256, 256, 0, stream>>>(W_out, Wout_t, DINNER, DINNER);
    k_transpose_bf16<<<(DINNER * DINNER + 255) / 256, 256, 0, stream>>>(W_ff1, Wff1_t, DINNER, DINNER);
    k_transpose_bf16<<<(DINNER * DINNER + 255) / 256, 256, 0, stream>>>(W_ff2, Wff2_t, DINNER, DINNER);

    k_build_xskv<<<(ROWS * LLM_EMB) / 256, 256, 0, stream>>>(vlm_keys, vlm_vals, Xskv);
    k_rmsnorm_bf16<<<ROWS, 256, 0, stream>>>(planning, g_norm1, normed1);

    // state_kv = X @ W_skv + b_skv     (4096 x 4096 x 512)
    k_wmma_gemm<<<dim3(DINNER / 64, ROWS / 128), 128, 0, stream>>>(
        Xskv, Wskv_t, b_skv, nullptr, statekv, ROWS, DINNER, LLM_EMB);
    // qkv = normed @ W_qkv + b_qkv     (4096 x 512 x 1536)
    k_wmma_gemm<<<dim3(3 * DINNER / 64, ROWS / 128), 128, 0, stream>>>(
        normed1, Wqkv_t, b_qkv, nullptr, qkvf, ROWS, 3 * DINNER, DINNER);

    k_rope_qk<<<(NB * HEADS * NPLAN * 32) / 256, 256, 0, stream>>>(qkvf, qbf, Kbf);
    k_fill_state_keys<<<(NB * HEADS * SVLM * DHEAD) / 256, 256, 0, stream>>>(statekv, Kbf);
    k_fill_vt<<<(NB * HEADS * KLEN * DHEAD) / 256, 256, 0, stream>>>(statekv, qkvf, Vtbf);

    k_attention<<<NB * HEADS * (NPLAN / 16), 32, 0, stream>>>(qbf, Kbf, Vtbf, attnbf);

    // all_out = attn @ W_out + b_out + planning
    k_wmma_gemm<<<dim3(DINNER / 64, ROWS / 128), 128, 0, stream>>>(
        attnbf, Wout_t, b_out, planning, allout, ROWS, DINNER, DINNER);

    k_rmsnorm_bf16<<<ROWS, 256, 0, stream>>>(allout, g_norm2, normed2);

    k_wmma_gemm<<<dim3(DINNER / 64, ROWS / 128), 128, 0, stream>>>(
        normed2, Wff1_t, b_ff1, nullptr, ffn1f, ROWS, DINNER, DINNER);
    k_gelu_bf16<<<(ROWS * DINNER) / 256, 256, 0, stream>>>(ffn1f, ffn1bf);
    // out = ffn1 @ W_ff2 + b_ff2 + all_out
    k_wmma_gemm<<<dim3(DINNER / 64, ROWS / 128), 128, 0, stream>>>(
        ffn1bf, Wff2_t, b_ff2, allout, out, ROWS, DINNER, DINNER);
}
